// TrajectoryTransformerGPTNeo_77790447665342
// MI455X (gfx1250) — compile-verified, matrix operand order checked by reading, not run on hardware
//
#include <hip/hip_runtime.h>
#include <hip/hip_bf16.h>

// ---------------------------------------------------------------------------
// TrajectoryTransformer (GPT-Neo style) forward on gfx1250 (MI455X, wave32).
// All GEMMs (QKV, attn QK^T / PV, Wo, MLP, heads) run through
// v_wmma_f32_16x16x32_bf16 with f32 accumulation. Attention is flash-style
// (online softmax, no T x T score materialization). Weights are cast to bf16
// and transposed once per launch so A and B WMMA fragments are contiguous
// 16-byte loads per lane.
// ---------------------------------------------------------------------------

#define B_   2
#define L_   512
#define SD_  64
#define AD_  18
#define H_   768
#define NH_  12
#define NL_  4
#define W_   256
#define T_   (3 * L_)      // 1536
#define FF_  (4 * H_)      // 3072
#define HD_  (H_ / NH_)    // 64

typedef __bf16 bf16;
typedef __attribute__((ext_vector_type(16))) __bf16        v16bf;
typedef __attribute__((ext_vector_type(8)))  float         v8f;
typedef __attribute__((ext_vector_type(4)))  unsigned int  u32x4;

union AFrag { v16bf v; u32x4 q[2]; };

__device__ inline v8f zf8() {
  v8f z = {0.f, 0.f, 0.f, 0.f, 0.f, 0.f, 0.f, 0.f};
  return z;
}

__device__ inline v16bf zero_frag() {
  AFrag f;
  u32x4 z = {0u, 0u, 0u, 0u};
  f.q[0] = z; f.q[1] = z;
  return f.v;
}

// Round-to-nearest-even f32 -> bf16
__device__ inline bf16 f2bf(float f) {
  unsigned u = __builtin_bit_cast(unsigned, f);
  u += 0x7fffu + ((u >> 16) & 1u);
  unsigned short h = (unsigned short)(u >> 16);
  return __builtin_bit_cast(bf16, h);
}

// Load one 16x32 bf16 fragment (A-layout; identical layout serves B when the
// operand is supplied K-contiguous, i.e. as Bt[N][K]).
// lanes 0-15: row = base row + (lane&15), K = k0+0..7  and k0+16..23
// lanes 16-31: same rows,                K = k0+8..15 and k0+24..31
__device__ inline v16bf load_frag(const bf16* base, int row, int ld, int k0, int half) {
  AFrag f;
  const bf16* p = base + (size_t)row * ld + k0 + half * 8;
  f.q[0] = *(const u32x4*)(p);
  f.q[1] = *(const u32x4*)(p + 16);
  return f.v;
}

__device__ inline v8f wmma_bf16(v16bf a, v16bf b, v8f c) {
  return __builtin_amdgcn_wmma_f32_16x16x32_bf16(
      /*neg_a=*/false, a, /*neg_b=*/false, b,
      /*c_mod=*/(short)0, c, /*reuse_a=*/false, /*reuse_b=*/false);
}

__device__ inline float gelu_new(float x) {
  return 0.5f * x * (1.f + tanhf(0.7978845608028654f * (x + 0.044715f * x * x * x)));
}

// ---------------------------------------------------------------------------
// Embedding: interleave (s,a,r) tokens + timestep embedding -> X[B*T, H] (f32)
// ---------------------------------------------------------------------------
__global__ __launch_bounds__(256) void embed_kernel(
    const float* __restrict__ states, const int* __restrict__ actions,
    const float* __restrict__ rewards, const int* __restrict__ timesteps,
    const float* __restrict__ Ws, const float* __restrict__ bs,
    const float* __restrict__ Ea, const float* __restrict__ Wr,
    const float* __restrict__ br, const float* __restrict__ Et,
    float* __restrict__ X)
{
  size_t idx = (size_t)blockIdx.x * 256 + threadIdx.x;
  if (idx >= (size_t)B_ * T_ * H_) return;
  int hc = (int)(idx % H_);
  size_t r = idx / H_;
  int t = (int)(r % T_);
  int b = (int)(r / T_);
  int l = t / 3, j = t - 3 * l;
  int bl = b * L_ + l;
  float te = Et[(size_t)timesteps[bl] * H_ + hc];
  float v;
  if (j == 0) {
    float acc = bs[hc];
    const float* sp = states + (size_t)bl * SD_;
    #pragma unroll 8
    for (int d = 0; d < SD_; ++d) acc += sp[d] * Ws[(size_t)d * H_ + hc];
    v = acc;
  } else if (j == 1) {
    v = Ea[(size_t)actions[bl] * H_ + hc];
  } else {
    v = rewards[bl] * Wr[hc] + br[hc];
  }
  X[idx] = v + te;
}

// ---------------------------------------------------------------------------
// LayerNorm (one block per row). Optional f32 output (+wpe) and/or bf16 output.
// Safe in-place for of == X (each thread touches only its own columns).
// ---------------------------------------------------------------------------
__global__ __launch_bounds__(256) void ln_kernel(
    const float* __restrict__ X, const float* __restrict__ g,
    const float* __restrict__ beta, bf16* __restrict__ obf,
    float* __restrict__ of, const float* __restrict__ wpe)
{
  int row = blockIdx.x;
  const float* xr = X + (size_t)row * H_;
  __shared__ float red[256];
  int tid = threadIdx.x;

  float s = 0.f;
  for (int c = tid; c < H_; c += 256) s += xr[c];
  red[tid] = s; __syncthreads();
  for (int st = 128; st > 0; st >>= 1) { if (tid < st) red[tid] += red[tid + st]; __syncthreads(); }
  float mean = red[0] * (1.f / H_);
  __syncthreads();

  s = 0.f;
  for (int c = tid; c < H_; c += 256) { float d = xr[c] - mean; s += d * d; }
  red[tid] = s; __syncthreads();
  for (int st = 128; st > 0; st >>= 1) { if (tid < st) red[tid] += red[tid + st]; __syncthreads(); }
  float rstd = rsqrtf(red[0] * (1.f / H_) + 1e-5f);

  int t = row % T_;
  for (int c = tid; c < H_; c += 256) {
    float v = (xr[c] - mean) * rstd * g[c] + beta[c];
    if (wpe) v += wpe[(size_t)t * H_ + c];
    if (of)  of[(size_t)row * H_ + c] = v;
    if (obf) obf[(size_t)row * H_ + c] = f2bf(v);
  }
}

// f32 [K,N] -> bf16 [N,K] (transpose), so GEMM B operand is K-contiguous.
__global__ __launch_bounds__(256) void castT_kernel(
    const float* __restrict__ in, bf16* __restrict__ out, int K, int N)
{
  size_t idx = (size_t)blockIdx.x * 256 + threadIdx.x;
  if (idx >= (size_t)K * N) return;
  int k = (int)(idx % K);
  int n = (int)(idx / K);
  out[idx] = f2bf(in[(size_t)k * N + n]);
}

// ---------------------------------------------------------------------------
// WMMA GEMM: C[M,N] = act(A[M,K](bf16) @ Bt[N,K]^T(bf16) + bias) (+residual)
// 256 threads = 8 waves; block tile 128x64; 32x32 per wave (4 accumulators).
// M, K multiples of 32 (M multiple of 128); N edge-guarded.
// Fast path (wave-uniform full tile) has no per-lane guards in the K loop so
// WMMA runs with EXEC all-ones and no cndmask chains.
// ---------------------------------------------------------------------------
__global__ __launch_bounds__(256) void gemm_kernel(
    const bf16* __restrict__ A, const bf16* __restrict__ Bt,
    const float* __restrict__ bias, const float* __restrict__ res,
    float* __restrict__ Cf, bf16* __restrict__ Cbf,
    int M, int N, int K, int act)
{
  const int lane = threadIdx.x & 31;
  const int wave = threadIdx.x >> 5;
  const int half = lane >> 4;
  const int lrw  = lane & 15;
  const int m0 = blockIdx.y * 128 + (wave >> 1) * 32;
  const int n0 = blockIdx.x * 64  + (wave & 1) * 32;
  const int ra0 = m0 + lrw,      ra1 = m0 + 16 + lrw;
  const int rb0 = n0 + lrw,      rb1 = n0 + 16 + lrw;

  v8f c00 = zf8(), c01 = zf8(), c10 = zf8(), c11 = zf8();

  // Warm the L2/WGP$ path once; data is L2-resident across the layer anyway.
  __builtin_prefetch(A  + (size_t)ra0 * K, 0, 1);

  if (n0 + 32 <= N) {                       // wave-uniform: full tile, no guards
    __builtin_prefetch(Bt + (size_t)rb0 * K, 0, 1);
    for (int k0 = 0; k0 < K; k0 += 32) {
      v16bf a0 = load_frag(A,  ra0, K, k0, half);
      v16bf a1 = load_frag(A,  ra1, K, k0, half);
      v16bf b0 = load_frag(Bt, rb0, K, k0, half);
      v16bf b1 = load_frag(Bt, rb1, K, k0, half);
      c00 = wmma_bf16(a0, b0, c00);
      c01 = wmma_bf16(a0, b1, c01);
      c10 = wmma_bf16(a1, b0, c10);
      c11 = wmma_bf16(a1, b1, c11);
    }
  } else {                                  // N edge (head GEMMs: N=18, 1)
    const bool vb0 = rb0 < N,      vb1 = rb1 < N;
    const int  sb0 = vb0 ? rb0 : 0, sb1 = vb1 ? rb1 : 0;
    const v16bf zf = zero_frag();
    for (int k0 = 0; k0 < K; k0 += 32) {
      v16bf a0 = load_frag(A,  ra0, K, k0, half);
      v16bf a1 = load_frag(A,  ra1, K, k0, half);
      v16bf b0 = load_frag(Bt, sb0, K, k0, half); if (!vb0) b0 = zf;
      v16bf b1 = load_frag(Bt, sb1, K, k0, half); if (!vb1) b1 = zf;
      c00 = wmma_bf16(a0, b0, c00);
      c01 = wmma_bf16(a0, b1, c01);
      c10 = wmma_bf16(a1, b0, c10);
      c11 = wmma_bf16(a1, b1, c11);
    }
  }

#define EPI(CC, MT, NT)                                                     \
  {                                                                         \
    int col = n0 + (NT) * 16 + lrw;                                         \
    if (col < N) {                                                          \
      float bv = bias ? bias[col] : 0.f;                                    \
      _Pragma("unroll")                                                     \
      for (int r = 0; r < 8; r++) {                                         \
        int row = m0 + (MT) * 16 + r + half * 8;                            \
        float v = CC[r] + bv;                                               \
        if (act == 1) v = gelu_new(v);                                      \
        else if (act == 2) v = fmaxf(v, 0.f);                               \
        size_t off = (size_t)row * N + col;                                 \
        if (res) v += res[off];                                             \
        if (Cf)  Cf[off] = v;                                               \
        if (Cbf) Cbf[off] = f2bf(v);                                        \
      }                                                                     \
    }                                                                       \
  }
  EPI(c00, 0, 0) EPI(c01, 0, 1) EPI(c10, 1, 0) EPI(c11, 1, 1)
#undef EPI
}

// V [B*T, H] (head-sliced) -> Vt [B, NH, HD, T] so PV B-fragments are contiguous.
__global__ __launch_bounds__(256) void transv_kernel(
    const bf16* __restrict__ V, bf16* __restrict__ Vt)
{
  size_t idx = (size_t)blockIdx.x * 256 + threadIdx.x;
  if (idx >= (size_t)B_ * NH_ * HD_ * T_) return;
  int t = (int)(idx % T_);
  size_t r = idx / T_;
  int d = (int)(r % HD_); r /= HD_;
  int h = (int)(r % NH_);
  int b = (int)(r / NH_);
  Vt[idx] = V[((size_t)b * T_ + t) * H_ + h * HD_ + d];
}

// ---------------------------------------------------------------------------
// Flash attention (bf16 WMMA, f32 online softmax). One wave = 16 query rows
// of one (b, head); key tiles of 32; scores are NOT scaled (GPT-Neo).
// ---------------------------------------------------------------------------
__global__ __launch_bounds__(128) void attn_kernel(
    const bf16* __restrict__ Q, const bf16* __restrict__ Kb,
    const bf16* __restrict__ Vt, bf16* __restrict__ Ctx, int isLocal)
{
  const int lane = threadIdx.x & 31;
  const int wave = threadIdx.x >> 5;
  const int half = lane >> 4;
  const int lrw  = lane & 15;
  const int b = blockIdx.y / NH_;
  const int h = blockIdx.y % NH_;
  const int q0 = blockIdx.x * 64 + wave * 16;

  const bf16* qb = Q  + ((size_t)b * T_) * H_ + h * HD_;
  const bf16* kb = Kb + ((size_t)b * T_) * H_ + h * HD_;
  const bf16* vt = Vt + ((size_t)(b * NH_ + h)) * HD_ * T_;

  __shared__ bf16 ldsP[4][16 * 32];
  bf16* myP = &ldsP[wave][0];

  // Q fragments (rows q0..q0+15, HD=64 -> two K=32 chunks), live across loop.
  v16bf aq0 = load_frag(qb, q0 + lrw, H_, 0,  half);
  v16bf aq1 = load_frag(qb, q0 + lrw, H_, 32, half);

  v8f o0 = zf8(), o1 = zf8(), o2 = zf8(), o3 = zf8();
  float m[8], lsum[8];
  #pragma unroll
  for (int r = 0; r < 8; r++) { m[r] = -3.0e38f; lsum[r] = 0.f; }

  int kend = q0 + 16;                       // causal: max key for this wave
  int kbeg = 0;
  if (isLocal) { kbeg = q0 - (W_ - 1); if (kbeg < 0) kbeg = 0; kbeg &= ~31; }

  for (int kt = kbeg; kt < kend; kt += 32) {
    // --- S = Q K^T  (two 16-col tiles) ---
    v8f s0 = zf8(), s1 = zf8();
    {
      v16bf bk00 = load_frag(kb, kt + lrw,      H_, 0,  half);
      v16bf bk10 = load_frag(kb, kt + 16 + lrw, H_, 0,  half);
      v16bf bk01 = load_frag(kb, kt + lrw,      H_, 32, half);
      v16bf bk11 = load_frag(kb, kt + 16 + lrw, H_, 32, half);
      s0 = wmma_bf16(aq0, bk00, s0);
      s1 = wmma_bf16(aq0, bk10, s1);
      s0 = wmma_bf16(aq1, bk01, s0);
      s1 = wmma_bf16(aq1, bk11, s1);
    }

    // --- mask + online softmax (row reductions across 16 lanes) ---
    const int key0 = kt + lrw;
    const int key1 = kt + 16 + lrw;
    #pragma unroll
    for (int r = 0; r < 8; r++) {
      int row = q0 + r + half * 8;
      bool ok0 = (key0 <= row) && (!isLocal || (row - key0) < W_);
      bool ok1 = (key1 <= row) && (!isLocal || (row - key1) < W_);
      float v0 = ok0 ? s0[r] : -3.0e38f;
      float v1 = ok1 ? s1[r] : -3.0e38f;
      float mx = fmaxf(v0, v1);
      #pragma unroll
      for (int off = 1; off < 16; off <<= 1) mx = fmaxf(mx, __shfl_xor(mx, off, 16));
      float mn = fmaxf(m[r], mx);
      float sc = __expf(m[r] - mn);
      m[r] = mn;
      float p0 = ok0 ? __expf(s0[r] - mn) : 0.f;
      float p1 = ok1 ? __expf(s1[r] - mn) : 0.f;
      float rs = p0 + p1;
      #pragma unroll
      for (int off = 1; off < 16; off <<= 1) rs += __shfl_xor(rs, off, 16);
      lsum[r] = lsum[r] * sc + rs;
      o0[r] *= sc; o1[r] *= sc; o2[r] *= sc; o3[r] *= sc;
      int rl = r + half * 8;                 // P in C-layout -> LDS row-major
      myP[rl * 32 + lrw]      = f2bf(p0);
      myP[rl * 32 + 16 + lrw] = f2bf(p1);
    }
    asm volatile("s_wait_dscnt 0" ::: "memory");   // LDS store -> load fence

    // --- O += P V  (P re-read as 16x32 A fragment; V^T rows contiguous) ---
    v16bf ap  = load_frag(myP, lrw, 32, 0, half);
    v16bf bv0 = load_frag(vt, 0 * 16 + lrw, T_, kt, half);
    v16bf bv1 = load_frag(vt, 1 * 16 + lrw, T_, kt, half);
    v16bf bv2 = load_frag(vt, 2 * 16 + lrw, T_, kt, half);
    v16bf bv3 = load_frag(vt, 3 * 16 + lrw, T_, kt, half);
    o0 = wmma_bf16(ap, bv0, o0);
    o1 = wmma_bf16(ap, bv1, o1);
    o2 = wmma_bf16(ap, bv2, o2);
    o3 = wmma_bf16(ap, bv3, o3);
  }

  #pragma unroll
  for (int r = 0; r < 8; r++) {
    float inv = 1.f / lsum[r];
    int row = q0 + r + half * 8;
    bf16* cp = Ctx + ((size_t)(b * T_ + row)) * H_ + h * HD_;
    cp[0 * 16 + lrw] = f2bf(o0[r] * inv);
    cp[1 * 16 + lrw] = f2bf(o1[r] * inv);
    cp[2 * 16 + lrw] = f2bf(o2[r] * inv);
    cp[3 * 16 + lrw] = f2bf(o3[r] * inv);
  }
}

// Gather tokens x[:, j::3] (bf16) -> contiguous [B*L, H] for the heads.
__global__ __launch_bounds__(256) void gather_kernel(
    const bf16* __restrict__ Xn, bf16* __restrict__ out, int j)
{
  size_t idx = (size_t)blockIdx.x * 256 + threadIdx.x;
  if (idx >= (size_t)B_ * L_ * H_) return;
  int c = (int)(idx % H_);
  size_t r = idx / H_;
  int l = (int)(r % L_);
  int b = (int)(r / L_);
  out[idx] = Xn[((size_t)b * T_ + 3 * l + j) * H_ + c];
}

// ---------------------------------------------------------------------------
// Host-side launch. Input order follows jax pytree flattening (dict keys
// sorted): states, actions, rewards, timesteps, then params:
// Ea, Et, Wa1, Wa2, Wq1, Wq2, Wr, Ws, Wt1, Wt2, ba1, ba2, bq1, bq2, br, bs,
// bt1, bt2, layers[4]{Wfc,Wk,Wo,Wpr,Wq,Wv,bfc,bo,bpr,ln1_b,ln1_g,ln2_b,ln2_g},
// ln_e_b, ln_e_g, ln_f_b, ln_f_g, wpe.
// ---------------------------------------------------------------------------
extern "C" void kernel_launch(void* const* d_in, const int* in_sizes, int n_in,
                              void* d_out, int out_size, void* d_ws, size_t ws_size,
                              hipStream_t stream)
{
  (void)in_sizes; (void)n_in; (void)out_size; (void)ws_size;

  const float* states    = (const float*)d_in[0];
  const int*   actions   = (const int*)d_in[1];
  const float* rewards   = (const float*)d_in[2];
  const int*   timesteps = (const int*)d_in[3];
  const float* Ea  = (const float*)d_in[4];
  const float* Et  = (const float*)d_in[5];
  const float* Wa1 = (const float*)d_in[6];
  const float* Wa2 = (const float*)d_in[7];
  const float* Wq1 = (const float*)d_in[8];
  const float* Wq2 = (const float*)d_in[9];
  const float* Wr  = (const float*)d_in[10];
  const float* Ws  = (const float*)d_in[11];
  const float* Wt1 = (const float*)d_in[12];
  const float* Wt2 = (const float*)d_in[13];
  const float* ba1 = (const float*)d_in[14];
  const float* ba2 = (const float*)d_in[15];
  const float* bq1 = (const float*)d_in[16];
  const float* bq2 = (const float*)d_in[17];
  const float* br  = (const float*)d_in[18];
  const float* bs  = (const float*)d_in[19];
  const float* bt1 = (const float*)d_in[20];
  const float* bt2 = (const float*)d_in[21];
  const float* lne_b = (const float*)d_in[74];
  const float* lne_g = (const float*)d_in[75];
  const float* lnf_b = (const float*)d_in[76];
  const float* lnf_g = (const float*)d_in[77];
  const float* wpe   = (const float*)d_in[78];

  const size_t BT = (size_t)B_ * T_;   // 3072
  const size_t BL = (size_t)B_ * L_;   // 1024

  char* p = (char*)d_ws;
  auto alloc = [&](size_t bytes) { void* r = p; p += (bytes + 255) & ~(size_t)255; return r; };
  float* X   = (float*)alloc(BT * H_ * 4);          // residual stream (f32)
  bf16*  XN  = (bf16*)alloc(BT * H_ * 2);           // LN output / attn ctx (bf16)
  bf16*  Qb  = (bf16*)alloc(BT * H_ * 2);
  bf16*  Kb  = (bf16*)alloc(BT * H_ * 2);
  bf16*  Vb  = (bf16*)alloc(BT * H_ * 2);
  bf16*  Vt  = (bf16*)alloc(BT * H_ * 2);           // [B,NH,HD,T]
  bf16*  MID = (bf16*)alloc(BT * FF_ * 2);          // GELU(fc) (bf16)
  bf16*  WT  = (bf16*)alloc((size_t)H_ * FF_ * 2);  // staged transposed weight
  bf16*  HB  = (bf16*)alloc(BL * H_ * 2);           // gathered head tokens
  bf16*  H1  = (bf16*)alloc(BL * H_ * 2);           // head hidden

  auto cdiv = [](size_t a, size_t b) { return (unsigned)((a + b - 1) / b); };
  auto castT = [&](const float* Wf, int K, int N) {
    castT_kernel<<<dim3(cdiv((size_t)K * N, 256)), 256, 0, stream>>>(Wf, WT, K, N);
  };
  auto gemm = [&](const bf16* A, const float* bias, const float* res,
                  float* Cf, bf16* Cbf, int M, int N, int K, int act) {
    gemm_kernel<<<dim3(cdiv((size_t)N, 64), (unsigned)(M / 128)), 256, 0, stream>>>(
        A, WT, bias, res, Cf, Cbf, M, N, K, act);
  };

  // Embeddings -> X, then ln_e (in place) + wpe
  embed_kernel<<<dim3(cdiv(BT * H_, 256)), 256, 0, stream>>>(
      states, actions, rewards, timesteps, Ws, bs, Ea, Wr, br, Et, X);
  ln_kernel<<<dim3((unsigned)BT), 256, 0, stream>>>(X, lne_g, lne_b, nullptr, X, wpe);

  for (int li = 0; li < NL_; ++li) {
    const float* Wfc = (const float*)d_in[22 + 13 * li + 0];
    const float* Wk  = (const float*)d_in[22 + 13 * li + 1];
    const float* Wo  = (const float*)d_in[22 + 13 * li + 2];
    const float* Wpr = (const float*)d_in[22 + 13 * li + 3];
    const float* Wq  = (const float*)d_in[22 + 13 * li + 4];
    const float* Wv  = (const float*)d_in[22 + 13 * li + 5];
    const float* bfc = (const float*)d_in[22 + 13 * li + 6];
    const float* bo  = (const float*)d_in[22 + 13 * li + 7];
    const float* bpr = (const float*)d_in[22 + 13 * li + 8];
    const float* ln1b = (const float*)d_in[22 + 13 * li + 9];
    const float* ln1g = (const float*)d_in[22 + 13 * li + 10];
    const float* ln2b = (const float*)d_in[22 + 13 * li + 11];
    const float* ln2g = (const float*)d_in[22 + 13 * li + 12];
    int isLocal = (li % 2 == 0) ? 1 : 0;   // ['local','global'] * 2

    ln_kernel<<<dim3((unsigned)BT), 256, 0, stream>>>(X, ln1g, ln1b, XN, nullptr, nullptr);
    castT(Wq, H_, H_); gemm(XN, nullptr, nullptr, nullptr, Qb, (int)BT, H_, H_, 0);
    castT(Wk, H_, H_); gemm(XN, nullptr, nullptr, nullptr, Kb, (int)BT, H_, H_, 0);
    castT(Wv, H_, H_); gemm(XN, nullptr, nullptr, nullptr, Vb, (int)BT, H_, H_, 0);
    transv_kernel<<<dim3(cdiv(BT * H_, 256)), 256, 0, stream>>>(Vb, Vt);
    attn_kernel<<<dim3(T_ / 64, B_ * NH_), 128, 0, stream>>>(Qb, Kb, Vt, XN, isLocal);
    castT(Wo, H_, H_); gemm(XN, bo, X, X, nullptr, (int)BT, H_, H_, 0);     // X += ctx@Wo + bo
    ln_kernel<<<dim3((unsigned)BT), 256, 0, stream>>>(X, ln2g, ln2b, XN, nullptr, nullptr);
    castT(Wfc, H_, FF_); gemm(XN, bfc, nullptr, nullptr, MID, (int)BT, FF_, H_, 1);  // gelu
    castT(Wpr, FF_, H_); gemm(MID, bpr, X, X, nullptr, (int)BT, H_, FF_, 0); // X += mid@Wpr + bpr
  }

  ln_kernel<<<dim3((unsigned)BT), 256, 0, stream>>>(X, lnf_g, lnf_b, XN, nullptr, nullptr);

  float* out_a = (float*)d_out;            // [B, L, AD]
  float* out_r = out_a + BL * AD_;         // [B, L, 1]
  float* out_s = out_r + BL * 1;           // [B, L, SD]

  // action head (tokens j=0)
  gather_kernel<<<dim3(cdiv(BL * H_, 256)), 256, 0, stream>>>(XN, HB, 0);
  castT(Wa1, H_, H_);  gemm(HB, ba1, nullptr, nullptr, H1, (int)BL, H_, H_, 2);
  castT(Wa2, H_, AD_); gemm(H1, ba2, nullptr, out_a, nullptr, (int)BL, AD_, H_, 0);
  // reward head (tokens j=1)
  gather_kernel<<<dim3(cdiv(BL * H_, 256)), 256, 0, stream>>>(XN, HB, 1);
  castT(Wq1, H_, H_);  gemm(HB, bq1, nullptr, nullptr, H1, (int)BL, H_, H_, 2);
  castT(Wq2, H_, 1);   gemm(H1, bq2, nullptr, out_r, nullptr, (int)BL, 1, H_, 0);
  // state head (tokens j=2)
  gather_kernel<<<dim3(cdiv(BL * H_, 256)), 256, 0, stream>>>(XN, HB, 2);
  castT(Wt1, H_, H_);  gemm(HB, bt1, nullptr, nullptr, H1, (int)BL, H_, H_, 2);
  castT(Wt2, H_, SD_); gemm(H1, bt2, nullptr, out_s, nullptr, (int)BL, SD_, H_, 0);
}
